// SparseMoELayer_47528108098236
// MI455X (gfx1250) — matrix operand order repaired and checked by reading, not compiled, not run on hardware
//
#include <hip/hip_runtime.h>
#include <hip/hip_bf16.h>

#define NUM_EXPERTS 8
#define INPUT_DIM   2048
#define OUTPUT_DIM  2048
#define TOP_K       2
#define TOKENS      4096

typedef __attribute__((ext_vector_type(16))) __bf16 v16bf;
typedef __attribute__((ext_vector_type(8)))  __bf16 v8bf;
typedef __attribute__((ext_vector_type(8)))  float  v8f;

static __device__ __forceinline__ __bf16 f2bf(float f) {
    // round-to-nearest-even fp32 -> bf16
    unsigned u = __builtin_bit_cast(unsigned, f);
    unsigned r = u + 0x7FFFu + ((u >> 16) & 1u);
    unsigned short s = (unsigned short)(r >> 16);
    return __builtin_bit_cast(__bf16, s);
}

// CDNA5 async copy: global -> LDS, 16B per lane, tracked by ASYNCcnt.
static __device__ __forceinline__ void async_copy_b128(unsigned lds_addr, const void* gptr) {
    asm volatile("global_load_async_to_lds_b128 %0, %1, off"
                 :: "v"(lds_addr), "v"(gptr) : "memory");
}
#define WAIT_ASYNCCNT_4() asm volatile("s_wait_asynccnt 4" ::: "memory")
#define WAIT_ASYNCCNT_0() asm volatile("s_wait_asynccnt 0" ::: "memory")

// ---------------------------------------------------------------------------
// Kernel 0: zero the output accumulator and per-expert counters.
// ---------------------------------------------------------------------------
__global__ void moe_zero_kernel(float* __restrict__ out, int n, int* __restrict__ cnt) {
    int i = blockIdx.x * blockDim.x + threadIdx.x;
    if (i < NUM_EXPERTS) cnt[i] = 0;
    for (int j = i; j < n; j += gridDim.x * blockDim.x) out[j] = 0.0f;
}

// ---------------------------------------------------------------------------
// Kernel 1: gating. One wave32 per token: 8 logits, top-2 softmax, atomic route.
// ---------------------------------------------------------------------------
__global__ void __launch_bounds__(256)
moe_gate_kernel(const float* __restrict__ x, const float* __restrict__ gw,
                const float* __restrict__ gb, int* __restrict__ cnt,
                int* __restrict__ tok, float* __restrict__ prob) {
    const int token = blockIdx.x * 8 + (threadIdx.x >> 5);
    const int lane  = threadIdx.x & 31;
    if (token >= TOKENS) return;

    float acc[NUM_EXPERTS];
#pragma unroll
    for (int e = 0; e < NUM_EXPERTS; ++e) acc[e] = 0.0f;

    const float* xr = x + (size_t)token * INPUT_DIM;
    for (int d = lane; d < INPUT_DIM; d += 32) {
        const float xv = xr[d];
        const float* g = gw + (size_t)d * NUM_EXPERTS;
#pragma unroll
        for (int e = 0; e < NUM_EXPERTS; ++e) acc[e] += xv * g[e];
    }
#pragma unroll
    for (int off = 16; off > 0; off >>= 1) {
#pragma unroll
        for (int e = 0; e < NUM_EXPERTS; ++e)
            acc[e] += __shfl_down(acc[e], off, 32);
    }
    if (lane == 0) {
        float logit[NUM_EXPERTS];
#pragma unroll
        for (int e = 0; e < NUM_EXPERTS; ++e) logit[e] = acc[e] + gb[e];
        int i0 = 0;
#pragma unroll
        for (int e = 1; e < NUM_EXPERTS; ++e) if (logit[e] > logit[i0]) i0 = e;
        int i1 = (i0 == 0) ? 1 : 0;
#pragma unroll
        for (int e = 0; e < NUM_EXPERTS; ++e)
            if (e != i0 && logit[e] > logit[i1]) i1 = e;
        const float e1  = __expf(logit[i1] - logit[i0]);
        const float inv = 1.0f / (1.0f + e1);

        int s0 = atomicAdd(&cnt[i0], 1);
        tok [i0 * TOKENS + s0] = token;
        prob[i0 * TOKENS + s0] = inv;
        int s1 = atomicAdd(&cnt[i1], 1);
        tok [i1 * TOKENS + s1] = token;
        prob[i1 * TOKENS + s1] = e1 * inv;
    }
}

// ---------------------------------------------------------------------------
// Kernel 2a: one-time fp32 -> bf16 conversion of x.
// ---------------------------------------------------------------------------
__global__ void __launch_bounds__(256)
conv_x_kernel(const float* __restrict__ x, __bf16* __restrict__ xb) {
    const size_t i = ((size_t)blockIdx.x * blockDim.x + threadIdx.x) * 8;
    v8bf o;
#pragma unroll
    for (int j = 0; j < 8; ++j) o[j] = f2bf(x[i + j]);
    *(v8bf*)(xb + i) = o;
}

// ---------------------------------------------------------------------------
// Kernel 2b: one-time fp32 -> bf16 conversion + transpose of expert weights:
//   W[e][k][n] (fp32) -> Wt[e][n][k] (bf16), so B fragments are contiguous.
// ---------------------------------------------------------------------------
__global__ void __launch_bounds__(256)
conv_wt_kernel(const float* __restrict__ w, __bf16* __restrict__ wt) {
    __shared__ __bf16 T[128][40];          // [n][k]
    const int e  = blockIdx.z;
    const int n0 = blockIdx.x * 128;
    const int k0 = blockIdx.y * 32;
    const int t  = threadIdx.x;
    {   // coalesced read: fixed k row, 16 contiguous n
        const int kk = t >> 3;             // 0..31
        const int nn = (t & 7) * 16;       // 0..112
        const float* src = w + ((size_t)e * INPUT_DIM + k0 + kk) * OUTPUT_DIM + n0 + nn;
#pragma unroll
        for (int i = 0; i < 16; ++i) T[nn + i][kk] = f2bf(src[i]);
    }
    __syncthreads();
    {   // coalesced write: fixed n row, 16 contiguous k
        const int n  = t >> 1;             // 0..127
        const int kk = (t & 1) * 16;       // 0 or 16
        __bf16* dst = wt + ((size_t)e * OUTPUT_DIM + n0 + n) * INPUT_DIM + k0 + kk;
        v8bf o0, o1;
#pragma unroll
        for (int i = 0; i < 8; ++i) { o0[i] = T[n][kk + i]; o1[i] = T[n][kk + 8 + i]; }
        *(v8bf*)(dst)     = o0;
        *(v8bf*)(dst + 8) = o1;
    }
}

// ---------------------------------------------------------------------------
// Kernel 3 (fast path): grouped GEMM, bf16 WMMA, fp32 accumulate.
// Block tile 128x128, K-step 64, 8 waves in a 4(M) x 2(N) grid.
// Each wave: 32 rows x 64 cols = 2 A-frags x 4 B-frags x 2 K-substeps
//  -> 16 WMMA per K-step, every B fragment reused by 2 WMMAs (ds:wmma = 1:1).
//  - A fragments: global_load_b128 straight from bf16 token rows (WGP$-hot).
//  - B tile: async global->LDS (ASYNCcnt), double buffered: next 16KB tile
//    streams in while the current tile's 16 WMMAs execute.
// ---------------------------------------------------------------------------
__global__ void __launch_bounds__(256)
moe_gemm_fast(const __bf16* __restrict__ xb, const __bf16* __restrict__ wt,
              const float* __restrict__ bias, const int* __restrict__ cnt,
              const int* __restrict__ tok, const float* __restrict__ prob,
              float* __restrict__ out) {
    const int e  = blockIdx.z;
    const int n0 = blockIdx.x * 128;
    const int m0 = blockIdx.y * 128;
    const int ce = cnt[e];
    if (m0 >= ce) return;                      // block-uniform: EXEC stays all-ones

    __shared__ __bf16 Bs[2][128][72];          // [buf][n][k], 144B row stride
    __shared__ int    tokLds[128];
    __shared__ float  probLds[128];

    const int t    = threadIdx.x;
    const int lane = t & 31;
    const int wave = t >> 5;
    const int wm   = wave & 3;                 // 0..3 -> rows [wm*32, +32)
    const int wn   = wave >> 2;                // 0..1 -> cols [wn*64, +64)

    if (t < 128) {
        const int r = m0 + t;
        const bool v = (r < ce);
        tokLds[t]  = v ? tok [e * TOKENS + r] : 0;
        probLds[t] = v ? prob[e * TOKENS + r] : 0.0f;
    }
    __syncthreads();

    const int g = lane >> 4;                   // lane half-group
    const int l15 = lane & 15;
    const __bf16* xrow0 = xb + (size_t)tokLds[wm * 32 + l15]      * INPUT_DIM;
    const __bf16* xrow1 = xb + (size_t)tokLds[wm * 32 + 16 + l15] * INPUT_DIM;
    const __bf16* we    = wt + (size_t)e * OUTPUT_DIM * INPUT_DIM;

    // per-thread B staging: 64B (4 x b128) of one Bs row per K-step
    const int sn = t >> 1;                     // 0..127
    const int sk = (t & 1) * 32;               // 0 or 32
    const __bf16* bsrc = we + (size_t)(n0 + sn) * INPUT_DIM + sk;

    v8f acc[8] = {};                           // [mi*4 + ni]

    auto issueB = [&](int i, int p) {
        const __bf16* s = bsrc + (size_t)i * 64;
        unsigned l = (unsigned)(size_t)&Bs[p][sn][sk];
        async_copy_b128(l,      s);
        async_copy_b128(l + 16, s + 8);
        async_copy_b128(l + 32, s + 16);
        async_copy_b128(l + 48, s + 24);
    };

    issueB(0, 0);                              // prologue: tile 0 -> buffer 0

    constexpr int NT = INPUT_DIM / 64;         // 32 K-steps
    for (int i = 0; i < NT; ++i) {
        const int p = i & 1;
        if (i + 1 < NT) {
            issueB(i + 1, 1 - p);              // stream next tile into other buffer
            WAIT_ASYNCCNT_4();                 // this wave's tile-i stores landed
        } else {
            WAIT_ASYNCCNT_0();
        }
        __syncthreads();                       // all waves' tile i landed

#pragma unroll
        for (int sub = 0; sub < 2; ++sub) {    // two K=32 sub-steps
            const int kofs = i * 64 + sub * 32;
            // A fragments for both 16-row strips of this wave
            const v16bf a0 = __builtin_shufflevector(
                *(const v8bf*)(xrow0 + kofs + g * 8),
                *(const v8bf*)(xrow0 + kofs + 16 + g * 8),
                0,1,2,3,4,5,6,7,8,9,10,11,12,13,14,15);
            const v16bf a1 = __builtin_shufflevector(
                *(const v8bf*)(xrow1 + kofs + g * 8),
                *(const v8bf*)(xrow1 + kofs + 16 + g * 8),
                0,1,2,3,4,5,6,7,8,9,10,11,12,13,14,15);
            const int kb = sub * 32 + g * 8;
#pragma unroll
            for (int ni = 0; ni < 4; ++ni) {
                const int n = wn * 64 + ni * 16 + l15;
                const v16bf b = __builtin_shufflevector(
                    *(const v8bf*)&Bs[p][n][kb],
                    *(const v8bf*)&Bs[p][n][kb + 16],
                    0,1,2,3,4,5,6,7,8,9,10,11,12,13,14,15);
                acc[ni] = __builtin_amdgcn_wmma_f32_16x16x32_bf16(
                    false, a0, false, b, (short)0, acc[ni], false, false);
                acc[4 + ni] = __builtin_amdgcn_wmma_f32_16x16x32_bf16(
                    false, a1, false, b, (short)0, acc[4 + ni], false, false);
            }
        }
        __syncthreads();                       // done reading buf p before overwrite
    }

    // epilogue: out[token] += prob * (acc + bias_e)
#pragma unroll
    for (int mi = 0; mi < 2; ++mi) {
#pragma unroll
        for (int ni = 0; ni < 4; ++ni) {
            const int n  = n0 + wn * 64 + ni * 16 + l15;
            const float bv = bias[e * OUTPUT_DIM + n];
#pragma unroll
            for (int j = 0; j < 8; ++j) {
                const int rl = wm * 32 + mi * 16 + g * 8 + j;
                if (m0 + rl < ce) {
                    atomicAdd(&out[(size_t)tokLds[rl] * OUTPUT_DIM + n],
                              probLds[rl] * (acc[mi * 4 + ni][j] + bv));
                }
            }
        }
    }
}

// ---------------------------------------------------------------------------
// Kernel 3 (fallback, small ws): fp32->bf16 conversion inside the GEMM loop.
// ---------------------------------------------------------------------------
__global__ void __launch_bounds__(256)
moe_gemm_fallback(const float* __restrict__ x, const float* __restrict__ w,
                  const float* __restrict__ bias, const int* __restrict__ cnt,
                  const int* __restrict__ tok, const float* __restrict__ prob,
                  float* __restrict__ out) {
    const int e  = blockIdx.z;
    const int n0 = blockIdx.x * 128;
    const int m0 = blockIdx.y * 128;
    const int ce = cnt[e];
    if (m0 >= ce) return;

    __shared__ __bf16 As[128][40];
    __shared__ __bf16 Bs[128][40];
    __shared__ int    tokLds[128];
    __shared__ float  probLds[128];

    const int t    = threadIdx.x;
    const int lane = t & 31;
    const int wave = t >> 5;

    if (t < 128) {
        const int r = m0 + t;
        const bool v = (r < ce);
        tokLds[t]  = v ? tok [e * TOKENS + r] : 0;
        probLds[t] = v ? prob[e * TOKENS + r] : 0.0f;
    }
    __syncthreads();

    v8f acc[8] = {};
    const float* we = w + (size_t)e * INPUT_DIM * OUTPUT_DIM;
    const int mRow = wave * 16 + (lane & 15);
    const int g    = lane >> 4;

    for (int k0 = 0; k0 < INPUT_DIM; k0 += 32) {
        {
            const int r  = t >> 1;
            const int kk = (t & 1) * 16;
            const float* src = x + (size_t)tokLds[r] * INPUT_DIM + k0 + kk;
#pragma unroll
            for (int i = 0; i < 16; ++i) As[r][kk + i] = f2bf(src[i]);
        }
        {
            const int kk = t >> 3;
            const int nn = (t & 7) * 16;
            const float* src = we + (size_t)(k0 + kk) * OUTPUT_DIM + n0 + nn;
#pragma unroll
            for (int i = 0; i < 16; ++i) Bs[nn + i][kk] = f2bf(src[i]);
        }
        __syncthreads();

        v16bf a;
#pragma unroll
        for (int h = 0; h < 8; ++h) {
            a[h]     = As[mRow][g * 16 + h];
            a[8 + h] = As[mRow][g * 16 + 8 + h];
        }
        const int kb = g * 16;
#pragma unroll
        for (int tt = 0; tt < 8; ++tt) {
            const int n = tt * 16 + (lane & 15);
            v16bf b;
#pragma unroll
            for (int h = 0; h < 16; ++h) b[h] = Bs[n][kb + h];
            acc[tt] = __builtin_amdgcn_wmma_f32_16x16x32_bf16(
                false, a, false, b, (short)0, acc[tt], false, false);
        }
        __syncthreads();
    }

#pragma unroll
    for (int tt = 0; tt < 8; ++tt) {
        const int n  = n0 + tt * 16 + (lane & 15);
        const float bv = bias[e * OUTPUT_DIM + n];
#pragma unroll
        for (int j = 0; j < 8; ++j) {
            const int rl = wave * 16 + g * 8 + j;
            if (m0 + rl < ce) {
                atomicAdd(&out[(size_t)tokLds[rl] * OUTPUT_DIM + n],
                          probLds[rl] * (acc[tt][j] + bv));
            }
        }
    }
}

// ---------------------------------------------------------------------------
extern "C" void kernel_launch(void* const* d_in, const int* in_sizes, int n_in,
                              void* d_out, int out_size, void* d_ws, size_t ws_size,
                              hipStream_t stream) {
    const float* x        = (const float*)d_in[0];
    const float* expert_w = (const float*)d_in[1];
    const float* expert_b = (const float*)d_in[2];
    const float* gate_w   = (const float*)d_in[3];
    const float* gate_b   = (const float*)d_in[4];
    float* out = (float*)d_out;

    // ws layout: [cnt 16 ints][tok E*T ints][prob E*T floats][xb bf16][wt bf16]
    int*   cnt  = (int*)d_ws;
    int*   tok  = cnt + 16;
    float* prob = (float*)(tok + NUM_EXPERTS * TOKENS);
    size_t off  = 64 + (size_t)NUM_EXPERTS * TOKENS * 8;
    off = (off + 255) & ~(size_t)255;
    __bf16* xb = (__bf16*)((char*)d_ws + off);
    size_t xb_bytes = (size_t)TOKENS * INPUT_DIM * 2;
    __bf16* wt = (__bf16*)((char*)xb + xb_bytes);
    size_t wt_bytes = (size_t)NUM_EXPERTS * INPUT_DIM * OUTPUT_DIM * 2;
    const bool fast = ws_size >= off + xb_bytes + wt_bytes;

    moe_zero_kernel<<<1024, 256, 0, stream>>>(out, TOKENS * OUTPUT_DIM, cnt);
    moe_gate_kernel<<<TOKENS / 8, 256, 0, stream>>>(x, gate_w, gate_b, cnt, tok, prob);

    dim3 grid(OUTPUT_DIM / 128, TOKENS / 128, NUM_EXPERTS);
    if (fast) {
        conv_x_kernel<<<(TOKENS * INPUT_DIM / 8) / 256, 256, 0, stream>>>(x, xb);
        dim3 cg(OUTPUT_DIM / 128, INPUT_DIM / 32, NUM_EXPERTS);
        conv_wt_kernel<<<cg, 256, 0, stream>>>(expert_w, wt);
        moe_gemm_fast<<<grid, 256, 0, stream>>>(xb, wt, expert_b, cnt, tok, prob, out);
    } else {
        moe_gemm_fallback<<<grid, 256, 0, stream>>>(x, expert_w, expert_b, cnt, tok, prob, out);
    }
}
